// TransformerDecoder_17575006175250
// MI455X (gfx1250) — matrix operand order repaired and checked by reading, not compiled, not run on hardware
//
#include <hip/hip_runtime.h>
#include <hip/hip_bf16.h>

// ---- problem constants (from reference) ----
#define L_   8
#define B_   4
#define D_   2048
#define HQ_  16
#define HKV_ 4
#define HD_  128
#define S_   2048
#define HID_ 4096
#define VT_  8192
#define VA_  1028
#define VACT_ 2048
#define LR_  256
#define NAUD_ 7
#define PAD_ 0
#define EPS_ 1e-5f
#define ROPE_BASE_ 10000.0f
#define GQ_  (HQ_ / HKV_)   // 4 heads per kv head

typedef __attribute__((ext_vector_type(16))) _Float16 v16h;
typedef __attribute__((ext_vector_type(8)))  float    v8f;

// ---------------------------------------------------------------------------
// 16 x (2x16) GEMM via v_wmma_f32_16x16x32_f16; two adjacent N-tiles per wave.
//   Y[m,n] = sum_k X[m,k] * W[k,n]  (+ R[m,n] if HASR),  for m < B_
// K and N are template constants so the 16 per-j B deltas are literal
// immediate offsets off one per-lane running pointer; HASR is a template
// constant so the epilogue is branch-free.
// Row tail on A and column tail on B are handled by CLAMPING indices:
// clamped lanes compute garbage that only lands in rows/columns never
// stored -> the streaming loop is completely branch-free.
// ---------------------------------------------------------------------------
template <int K, int N, bool HASR>
__global__ void __launch_bounds__(32)
gemm32_wmma(const float* __restrict__ X,
            const float* __restrict__ W,
            const float* __restrict__ R,   // residual (read iff HASR)
            float* __restrict__ Y)
{
    const int lane   = threadIdx.x & 31;
    const int n0     = blockIdx.x * 32;
    const int mrow   = lane & 15;                         // A row this lane carries
    const int mclamp = (mrow < B_) ? mrow : (B_ - 1);     // clamped (rows >= B_ unused)
    const int khalfA = (lane >> 4) * 16;                  // A K sub-offset
    const int kbaseB = (lane >> 4) * 16;                  // B K sub-offset
    const int col    = lane & 15;

    const int nc0 = n0 + col;                             // tile-0 column
    const int nc1 = n0 + 16 + col;                        // tile-1 column
    const int ncl0 = (nc0 < N) ? nc0 : (N - 1);           // clamped load columns
    const int ncl1 = (nc1 < N) ? nc1 : (N - 1);

    // per-lane running pointers (advance by constant strides each chunk)
    const float* xr  = X + (size_t)mclamp * K + khalfA;
    const float* wp0 = W + (size_t)kbaseB * N + ncl0;
    const float* wp1 = W + (size_t)kbaseB * N + ncl1;

    v8f acc0 = {}, acc1 = {};
    for (int k0 = 0; k0 < K; k0 += 32) {
        // ---- A tile: 4 x b128 with literal offsets ----
        v16h a;
#pragma unroll
        for (int j = 0; j < 8; ++j) {
            a[j]     = (_Float16)xr[j];
            a[8 + j] = (_Float16)xr[8 + j];
        }
        // ---- B tiles: 32 x b32 with literal j*N immediate offsets ----
        v16h b0, b1;
#pragma unroll
        for (int j = 0; j < 16; ++j) {
            b0[j] = (_Float16)wp0[j * N];
            b1[j] = (_Float16)wp1[j * N];
        }
        // prefetch next K-chunk of W into GL2 (global_prefetch_b8)
        __builtin_prefetch(wp0 + (size_t)32 * N, 0, 0);

        acc0 = __builtin_amdgcn_wmma_f32_16x16x32_f16(
                   false, a, false, b0, (short)0, acc0, false, false);
        acc1 = __builtin_amdgcn_wmma_f32_16x16x32_f16(
                   false, a, false, b1, (short)0, acc1, false, false);

        xr  += 32;
        wp0 += (size_t)32 * N;
        wp1 += (size_t)32 * N;
    }

    // ---- store rows < B_ (only lanes 0-15 carry rows 0..7) ----
    const int mbase = (lane >> 4) * 8;
#pragma unroll
    for (int i = 0; i < 8; ++i) {
        int m = mbase + i;
        if (m < B_) {
            if (nc0 < N) {
                float v = acc0[i];
                if constexpr (HASR) v += R[(size_t)m * N + nc0];
                Y[(size_t)m * N + nc0] = v;
            }
            if (nc1 < N) {
                float v = acc1[i];
                if constexpr (HASR) v += R[(size_t)m * N + nc1];
                Y[(size_t)m * N + nc1] = v;
            }
        }
    }
}

// ---------------------------------------------------------------------------
// Embedding: h[b,d] = (e1[t0]*m0 + e2[t1]*m1) @ text_up + sum_a audio[a, tok_a]
// ---------------------------------------------------------------------------
__global__ void embed_kernel(const int* __restrict__ tokens,
                             const float* __restrict__ e1,
                             const float* __restrict__ e2,
                             const float* __restrict__ up,
                             const float* __restrict__ audio,
                             float* __restrict__ h)
{
    int idx = blockIdx.x * blockDim.x + threadIdx.x;
    if (idx >= B_ * D_) return;
    int b = idx / D_, d = idx % D_;
    int t0 = tokens[b * 9 + 0];
    int t1 = tokens[b * 9 + 1];
    float m0 = (t0 != PAD_) ? 1.0f : 0.0f;
    float m1 = (t1 != PAD_) ? 1.0f : 0.0f;
    float acc = 0.0f;
    const float* r1 = e1 + (size_t)t0 * LR_;
    const float* r2 = e2 + (size_t)t1 * LR_;
    for (int r = 0; r < LR_; ++r) {
        float e = r1[r] * m0 + r2[r] * m1;
        acc = fmaf(e, up[(size_t)r * D_ + d], acc);
    }
    for (int a = 0; a < NAUD_; ++a) {
        int tok = tokens[b * 9 + 2 + a];
        acc += audio[((size_t)a * VA_ + tok) * D_ + d];
    }
    h[idx] = acc;
}

// ---------------------------------------------------------------------------
// RMSNorm: one block per batch row
// ---------------------------------------------------------------------------
__global__ void rmsnorm_kernel(const float* __restrict__ x,
                               const float* __restrict__ w,
                               float* __restrict__ y, int Dlen)
{
    __shared__ float red[256];
    int b = blockIdx.x;
    const float* xr = x + (size_t)b * Dlen;
    float s = 0.0f;
    for (int d = threadIdx.x; d < Dlen; d += blockDim.x) {
        float v = xr[d];
        s = fmaf(v, v, s);
    }
    red[threadIdx.x] = s;
    __syncthreads();
    for (int off = 128; off > 0; off >>= 1) {
        if ((int)threadIdx.x < off) red[threadIdx.x] += red[threadIdx.x + off];
        __syncthreads();
    }
    float inv = rsqrtf(red[0] / (float)Dlen + EPS_);
    float* yr = y + (size_t)b * Dlen;
    for (int d = threadIdx.x; d < Dlen; d += blockDim.x)
        yr[d] = xr[d] * inv * w[d];
}

// ---------------------------------------------------------------------------
// RoPE in place on q (B,HQ,HD) and k (B,HKV,HD)
// ---------------------------------------------------------------------------
__global__ void rope_kernel(float* __restrict__ q, float* __restrict__ k,
                            const int* __restrict__ positions)
{
    const int half = HD_ / 2;
    int idx = blockIdx.x * blockDim.x + threadIdx.x;
    int per_b = (HQ_ + HKV_) * half;
    if (idx >= B_ * per_b) return;
    int b = idx / per_b;
    int rem = idx % per_b;
    int h = rem / half;
    int i = rem % half;
    float* base = (h < HQ_) ? (q + ((size_t)b * HQ_ + h) * HD_)
                            : (k + ((size_t)b * HKV_ + (h - HQ_)) * HD_);
    float inv = __expf(-__logf(ROPE_BASE_) * (float)i / (float)half);
    float ang = (float)positions[b] * inv;
    float c = __cosf(ang), s = __sinf(ang);
    float x1 = base[i], x2 = base[i + half];
    base[i]        = x1 * c - x2 * s;
    base[i + half] = x1 * s + x2 * c;
}

// ---------------------------------------------------------------------------
// Attention for one (b, kv-head): GQ_=4 query heads share one K/V stream.
// Phase 1: one wave per position s; float4 K-row loads (512B/wave coalesced),
//          __shfl_xor wave reduction for the 4 grouped-head dots.
// Phase 2: softmax, ALL 4 heads in parallel (64 threads per head).
// Phase 3: P.V with float4 V loads; 4 heads x 2 sequence-halves on the
//          8 waves, halves combined through LDS.  s==pos rows come from the
//          freshly roped k/v (input caches are never mutated).
// ---------------------------------------------------------------------------
__global__ void __launch_bounds__(256)
attn_kernel(const float* __restrict__ q,
            const float* __restrict__ knew,
            const float* __restrict__ vnew,
            const float* __restrict__ cacheK,   // layer base: (B,HKV,S,HD)
            const float* __restrict__ cacheV,
            const int* __restrict__ positions,
            float* __restrict__ out)            // (B,HQ,HD)
{
    __shared__ float sc[GQ_][S_];               // 32 KB
    __shared__ float red[256];                  // 1 KB
    __shared__ float ored[2][GQ_][HD_];         // 4 KB

    const int b = blockIdx.x / HKV_;
    const int g = blockIdx.x % HKV_;
    const int pos = positions[b];
    const int lane = threadIdx.x & 31;
    const int wid  = threadIdx.x >> 5;     // 0..7
    const float scale = rsqrtf((float)HD_);

    const float* Kb = cacheK + ((size_t)b * HKV_ + g) * S_ * HD_;
    const float* Vb = cacheV + ((size_t)b * HKV_ + g) * S_ * HD_;
    const float* qg = q + ((size_t)b * HQ_ + g * GQ_) * HD_;
    const float* kN = knew + ((size_t)b * HKV_ + g) * HD_;
    const float* vN = vnew + ((size_t)b * HKV_ + g) * HD_;

    // preload q fragments: 4 heads x float4 at dim lane*4
    float4 qf[GQ_];
#pragma unroll
    for (int j = 0; j < GQ_; ++j)
        qf[j] = *(const float4*)(qg + (size_t)j * HD_ + lane * 4);

    // ---- phase 1: scores (one wave per s) ----
    for (int s = wid; s <= pos; s += 8) {
        const float* krow = (s == pos) ? kN : (Kb + (size_t)s * HD_);
        float4 kv = *(const float4*)(krow + lane * 4);
        float d0 = fmaf(qf[0].x, kv.x, fmaf(qf[0].y, kv.y, fmaf(qf[0].z, kv.z, qf[0].w * kv.w)));
        float d1 = fmaf(qf[1].x, kv.x, fmaf(qf[1].y, kv.y, fmaf(qf[1].z, kv.z, qf[1].w * kv.w)));
        float d2 = fmaf(qf[2].x, kv.x, fmaf(qf[2].y, kv.y, fmaf(qf[2].z, kv.z, qf[2].w * kv.w)));
        float d3 = fmaf(qf[3].x, kv.x, fmaf(qf[3].y, kv.y, fmaf(qf[3].z, kv.z, qf[3].w * kv.w)));
#pragma unroll
        for (int off = 16; off > 0; off >>= 1) {
            d0 += __shfl_xor(d0, off, 32);
            d1 += __shfl_xor(d1, off, 32);
            d2 += __shfl_xor(d2, off, 32);
            d3 += __shfl_xor(d3, off, 32);
        }
        if (lane == 0) {
            sc[0][s] = d0 * scale;
            sc[1][s] = d1 * scale;
            sc[2][s] = d2 * scale;
            sc[3][s] = d3 * scale;
        }
    }
    __syncthreads();

    // ---- phase 2: softmax, 4 heads in parallel (64 threads each) ----
    {
        const int hg = threadIdx.x >> 6;       // head 0..3
        const int ht = threadIdx.x & 63;       // thread within head group
        const int hb = hg << 6;                // group base in red[]

        float mx = -1e30f;
        for (int s = ht; s <= pos; s += 64) mx = fmaxf(mx, sc[hg][s]);
        red[threadIdx.x] = mx;
        __syncthreads();
        for (int off = 32; off > 0; off >>= 1) {
            if (ht < off)
                red[threadIdx.x] = fmaxf(red[threadIdx.x], red[threadIdx.x + off]);
            __syncthreads();
        }
        mx = red[hb];
        __syncthreads();

        float sum = 0.f;
        for (int s = ht; s <= pos; s += 64) {
            float e = __expf(sc[hg][s] - mx);
            sc[hg][s] = e;
            sum += e;
        }
        red[threadIdx.x] = sum;
        __syncthreads();
        for (int off = 32; off > 0; off >>= 1) {
            if (ht < off) red[threadIdx.x] += red[threadIdx.x + off];
            __syncthreads();
        }
        float rinv = 1.0f / red[hb];
        __syncthreads();
        for (int s = ht; s <= pos; s += 64) sc[hg][s] *= rinv;
        __syncthreads();
    }

    // ---- phase 3: out[h,:] = sum_s p[h,s] * V[s,:]  (float4 per lane) ----
    {
        const int h  = wid & 3;                // head
        const int sh = wid >> 2;               // sequence half 0/1
        float4 o = make_float4(0.f, 0.f, 0.f, 0.f);
        for (int s = sh; s < pos; s += 2) {
            float4 vv = *(const float4*)(Vb + (size_t)s * HD_ + lane * 4);
            float p = sc[h][s];
            o.x = fmaf(p, vv.x, o.x);
            o.y = fmaf(p, vv.y, o.y);
            o.z = fmaf(p, vv.z, o.z);
            o.w = fmaf(p, vv.w, o.w);
        }
        if ((pos & 1) == sh) {                 // s == pos row from new v
            float4 vv = *(const float4*)(vN + lane * 4);
            float p = sc[h][pos];
            o.x = fmaf(p, vv.x, o.x);
            o.y = fmaf(p, vv.y, o.y);
            o.z = fmaf(p, vv.z, o.z);
            o.w = fmaf(p, vv.w, o.w);
        }
        *(float4*)&ored[sh][h][lane * 4] = o;
        __syncthreads();
        if (sh == 0) {
            float4 a = *(const float4*)&ored[0][h][lane * 4];
            float4 c = *(const float4*)&ored[1][h][lane * 4];
            float4 r = make_float4(a.x + c.x, a.y + c.y, a.z + c.z, a.w + c.w);
            *(float4*)(out + ((size_t)b * HQ_ + (g * GQ_ + h)) * HD_ + lane * 4) = r;
        }
    }
}

// ---------------------------------------------------------------------------
// act[b,i] = silu(gu[b,i]) * gu[b,HID+i]
// ---------------------------------------------------------------------------
__global__ void silu_kernel(const float* __restrict__ gu, float* __restrict__ act)
{
    int idx = blockIdx.x * blockDim.x + threadIdx.x;
    if (idx >= B_ * HID_) return;
    int b = idx / HID_, i = idx % HID_;
    float g = gu[(size_t)b * 2 * HID_ + i];
    float u = gu[(size_t)b * 2 * HID_ + HID_ + i];
    float s = g / (1.0f + __expf(-g));
    act[idx] = s * u;
}

// ---------------------------------------------------------------------------
extern "C" void kernel_launch(void* const* d_in, const int* in_sizes, int n_in,
                              void* d_out, int out_size, void* d_ws, size_t ws_size,
                              hipStream_t stream)
{
    const int*   tokens    = (const int*)d_in[0];
    const int*   positions = (const int*)d_in[1];
    const float* cache_k   = (const float*)d_in[2];
    const float* cache_v   = (const float*)d_in[3];
    const float* text_emb1 = (const float*)d_in[4];
    const float* text_emb2 = (const float*)d_in[5];
    const float* text_up   = (const float*)d_in[6];
    const float* audio_tab = (const float*)d_in[7];
    const float* pre_w     = (const float*)d_in[8];
    const float* post_w    = (const float*)d_in[9];
    const float* Wq        = (const float*)d_in[10];
    const float* Wk        = (const float*)d_in[11];
    const float* Wv        = (const float*)d_in[12];
    const float* Wo        = (const float*)d_in[13];
    const float* Wi        = (const float*)d_in[14];
    const float* Wm        = (const float*)d_in[15];
    const float* final_w   = (const float*)d_in[16];
    const float* Wa        = (const float*)d_in[17];
    const float* Wc        = (const float*)d_in[18];

    // workspace layout (floats)
    float* ws  = (float*)d_ws;
    float* x   = ws;                    // B*D
    float* xn  = x   + B_ * D_;         // B*D
    float* q   = xn  + B_ * D_;         // B*HQ*HD
    float* k   = q   + B_ * HQ_ * HD_;  // B*HKV*HD
    float* v   = k   + B_ * HKV_ * HD_; // B*HKV*HD
    float* ao  = v   + B_ * HKV_ * HD_; // B*HQ*HD
    float* gu  = ao  + B_ * HQ_ * HD_;  // B*2*HID
    float* act = gu  + B_ * 2 * HID_;   // B*HID

    embed_kernel<<<(B_ * D_ + 255) / 256, 256, 0, stream>>>(
        tokens, text_emb1, text_emb2, text_up, audio_tab, x);

    const size_t cacheLayerStride = (size_t)B_ * HKV_ * S_ * HD_;
    for (int l = 0; l < L_; ++l) {
        const float* wq = Wq + (size_t)l * D_ * (HQ_ * HD_);
        const float* wk = Wk + (size_t)l * D_ * (HKV_ * HD_);
        const float* wv = Wv + (size_t)l * D_ * (HKV_ * HD_);
        const float* wo = Wo + (size_t)l * (HQ_ * HD_) * D_;
        const float* wi = Wi + (size_t)l * D_ * (2 * HID_);
        const float* wm = Wm + (size_t)l * HID_ * D_;
        const float* ckl = cache_k + (size_t)l * cacheLayerStride;
        const float* cvl = cache_v + (size_t)l * cacheLayerStride;

        rmsnorm_kernel<<<B_, 256, 0, stream>>>(x, pre_w + (size_t)l * D_, xn, D_);

        gemm32_wmma<D_, HQ_ * HD_, false><<<(HQ_ * HD_) / 32, 32, 0, stream>>>(
            xn, wq, nullptr, q);
        gemm32_wmma<D_, HKV_ * HD_, false><<<(HKV_ * HD_) / 32, 32, 0, stream>>>(
            xn, wk, nullptr, k);
        gemm32_wmma<D_, HKV_ * HD_, false><<<(HKV_ * HD_) / 32, 32, 0, stream>>>(
            xn, wv, nullptr, v);

        rope_kernel<<<(B_ * (HQ_ + HKV_) * (HD_ / 2) + 255) / 256, 256, 0, stream>>>(
            q, k, positions);

        attn_kernel<<<B_ * HKV_, 256, 0, stream>>>(q, k, v, ckl, cvl, positions, ao);

        // x = x + ao @ Wo
        gemm32_wmma<HQ_ * HD_, D_, true><<<D_ / 32, 32, 0, stream>>>(ao, wo, x, x);

        rmsnorm_kernel<<<B_, 256, 0, stream>>>(x, post_w + (size_t)l * D_, xn, D_);

        gemm32_wmma<D_, 2 * HID_, false><<<(2 * HID_) / 32, 32, 0, stream>>>(
            xn, wi, nullptr, gu);
        silu_kernel<<<(B_ * HID_ + 255) / 256, 256, 0, stream>>>(gu, act);

        // x = x + act @ Wm
        gemm32_wmma<HID_, D_, true><<<D_ / 32, 32, 0, stream>>>(act, wm, x, x);
    }

    float* hidden = (float*)d_out;                 // B*D
    float* alog   = hidden + B_ * D_;              // B*VACT
    float* clog   = alog   + B_ * VACT_;           // B*VA

    rmsnorm_kernel<<<B_, 256, 0, stream>>>(x, final_w, hidden, D_);
    gemm32_wmma<D_, VACT_, false><<<VACT_ / 32, 32, 0, stream>>>(
        hidden, Wa, nullptr, alog);
    gemm32_wmma<D_, VA_, false><<<(VA_ + 31) / 32, 32, 0, stream>>>(
        hidden, Wc, nullptr, clog);
}
